// HopfieldDQN_26173530702353
// MI455X (gfx1250) — matrix operand order repaired and checked by reading, not compiled, not run on hardware
//
#include <hip/hip_runtime.h>
#include <stdint.h>

#define BATCH    32768
#define IN_DIM   512
#define HID_DIM  1024
#define OUT_DIM  2
#define E_DIM    64
#define NHOP_IT  10
#define COMB_DIM (IN_DIM + E_DIM)   // 576

typedef __attribute__((ext_vector_type(16))) __bf16 v16bf;
typedef __attribute__((ext_vector_type(8)))  __bf16 v8bf;
typedef __attribute__((ext_vector_type(8)))  float  v8f;

__device__ __forceinline__ __bf16 f2bf(float f) {
  union { float f; uint32_t u; } v; v.f = f;
  uint32_t r = v.u + 0x7FFFu + ((v.u >> 16) & 1u);   // round-to-nearest-even
  uint16_t h = (uint16_t)(r >> 16);
  __bf16 b; __builtin_memcpy(&b, &h, 2); return b;
}
__device__ __forceinline__ float bf2f(__bf16 b) {
  uint16_t h; __builtin_memcpy(&h, &b, 2);
  union { uint32_t u; float f; } v; v.u = ((uint32_t)h) << 16; return v.f;
}

// CDNA5 async global->LDS copy (16B per lane), tracked by ASYNCcnt.
__device__ __forceinline__ void async_copy_b128(uint32_t lds_off, const __bf16* gptr) {
  asm volatile("global_load_async_to_lds_b128 %0, %1, off"
               :: "v"(lds_off), "v"(gptr) : "memory");
}
__device__ __forceinline__ void wait_asynccnt0() {
  asm volatile("s_wait_asynccnt 0x0" ::: "memory");
}

// ---------------- elementwise converts ----------------
__global__ void cvt_kernel(const float* __restrict__ src, __bf16* __restrict__ dst, int n) {
  int i = blockIdx.x * blockDim.x + threadIdx.x;
  if (i < n) dst[i] = f2bf(src[i]);
}

// transpose + convert: src [K,N] fp32 row-major -> dst [N,K] bf16 row-major
__global__ void cvt_t_kernel(const float* __restrict__ src, __bf16* __restrict__ dst,
                             int K, int N) {
  int i = blockIdx.x * blockDim.x + threadIdx.x;
  if (i < K * N) {
    int k = i / N, n = i % N;
    dst[(size_t)n * K + k] = f2bf(src[i]);
  }
}

// x [B,512] fp32 -> combined [B,576] bf16 (cols 0..511)
__global__ void x_pack_kernel(const float* __restrict__ x, __bf16* __restrict__ comb) {
  int i = blockIdx.x * blockDim.x + threadIdx.x;  // over B*512
  int row = i >> 9, col = i & 511;
  comb[(size_t)row * COMB_DIM + col] = f2bf(x[i]);
}

// ---------------- tiled bf16 WMMA GEMM: C = act(A @ B + bias) ----------------
// A: [M,K] bf16 row-major.  B is supplied PRE-TRANSPOSED: BwT [N,K] bf16
// row-major, which is exactly the LDS BsT[n][k] layout -> both A and B tiles
// are staged with global_load_async_to_lds_b128 (no VGPR staging, no ds
// stores in the hot loop). Double-buffered LDS; next tile's async DMA is in
// flight during the WMMA chain; fenced by s_wait_asynccnt + barrier.
// Wave tile = (16*MT) x (16*NT); block = (16*MT*MWAVES) x (16*NT*NWAVES); BK=32.
template<int MT, int NT, int MWAVES, int NWAVES, bool RELU, bool OUT_BF16>
__launch_bounds__(256)
__global__ void gemm_bf16_wmma(const __bf16* __restrict__ A, int lda,
                               const __bf16* __restrict__ BwT, int ldbt,
                               const float* __restrict__ bias,
                               void* __restrict__ C, int ldc, int K)
{
  constexpr int BM = 16 * MT * MWAVES;
  constexpr int BN = 16 * NT * NWAVES;
  constexpr int LDS_STRIDE = 40;           // 32 + 8 pad (banks)
  constexpr int ACH = BM * 4 / 256;        // 16B async chunks per thread (A tile)
  constexpr int BCH = BN * 4 / 256;        // 16B async chunks per thread (B tile)

  __shared__ __bf16 As[2][BM][LDS_STRIDE];
  __shared__ __bf16 BsT[2][BN][LDS_STRIDE];   // BsT[n][k]

  const int tid   = threadIdx.x;
  const int wave  = tid >> 5;
  const int lane  = tid & 31;
  const int nIdx  = lane & 15;   // N (B/C frags) / M (A frag)
  const int kh    = lane >> 4;   // half-select per ISA layouts
  const int mwave = wave / NWAVES;
  const int nwave = wave % NWAVES;
  const int rowBase = blockIdx.x * BM;
  const int colBase = blockIdx.y * BN;
  const int wRow = mwave * 16 * MT;
  const int wCol = nwave * 16 * NT;

  // accumulators init = bias (C/D layout: VGPR r -> M = 8*kh + r, N = nIdx)
  v8f acc[MT][NT];
  #pragma unroll
  for (int t = 0; t < NT; ++t) {
    float bv = bias[colBase + wCol + 16 * t + nIdx];
    #pragma unroll
    for (int mi = 0; mi < MT; ++mi)
      #pragma unroll
      for (int r = 0; r < 8; ++r) acc[mi][t][r] = bv;
  }

  auto loadA_async = [&](int buf, int k0) {
    #pragma unroll
    for (int c = 0; c < ACH; ++c) {
      int idx = tid + c * 256;
      int r = idx >> 2;            // A-tile row
      int ch = idx & 3;            // which 8-elem chunk of the 32-wide slab
      uint32_t lo = (uint32_t)(uintptr_t)&As[buf][r][ch * 8];
      const __bf16* gp = A + (size_t)(rowBase + r) * lda + (k0 + ch * 8);
      async_copy_b128(lo, gp);
    }
  };
  auto loadB_async = [&](int buf, int k0) {
    #pragma unroll
    for (int c = 0; c < BCH; ++c) {
      int idx = tid + c * 256;
      int n = idx >> 2;            // B-tile (transposed) row = output col
      int ch = idx & 3;
      uint32_t lo = (uint32_t)(uintptr_t)&BsT[buf][n][ch * 8];
      const __bf16* gp = BwT + (size_t)(colBase + n) * ldbt + (k0 + ch * 8);
      async_copy_b128(lo, gp);
    }
  };
  auto compute = [&](int buf) {
    v16bf af[MT], bfr[NT];
    #pragma unroll
    for (int mi = 0; mi < MT; ++mi) {
      // A frag (16-bit A 16x32): lane<16 -> K0..7|16..23 ; lane>=16 -> K8..15|24..31
      const __bf16* ap = &As[buf][wRow + 16 * mi + nIdx][kh * 8];
      v8bf lo = *(const v8bf*)(ap);
      v8bf hi = *(const v8bf*)(ap + 16);
      af[mi] = __builtin_shufflevector(lo, hi, 0,1,2,3,4,5,6,7,8,9,10,11,12,13,14,15);
    }
    #pragma unroll
    for (int t = 0; t < NT; ++t) {
      // B frag (Kx16): lane<16 -> K0..7|16..23 of col N ; lane>=16 -> K8..15|24..31
      const __bf16* bp = &BsT[buf][wCol + 16 * t + nIdx][kh * 8];
      v8bf lo = *(const v8bf*)(bp);
      v8bf hi = *(const v8bf*)(bp + 16);
      bfr[t] = __builtin_shufflevector(lo, hi, 0,1,2,3,4,5,6,7,8,9,10,11,12,13,14,15);
    }
    #pragma unroll
    for (int mi = 0; mi < MT; ++mi)
      #pragma unroll
      for (int t = 0; t < NT; ++t)
        acc[mi][t] = __builtin_amdgcn_wmma_f32_16x16x32_bf16(
            false, af[mi], false, bfr[t], (short)0, acc[mi][t], false, false);
  };

  const int KSTEPS = K / 32;

  // prologue: stage tile 0 via async DMA
  loadA_async(0, 0);
  loadB_async(0, 0);
  wait_asynccnt0();
  __syncthreads();

  int cur = 0;
  for (int s = 0; s < KSTEPS; ++s) {
    const int nxt = cur ^ 1;
    if (s + 1 < KSTEPS) {             // next tile's DMA in flight during WMMA
      loadA_async(nxt, 32 * (s + 1));
      loadB_async(nxt, 32 * (s + 1));
    }
    compute(cur);
    wait_asynccnt0();
    __syncthreads();
    cur = nxt;
  }

  #pragma unroll
  for (int mi = 0; mi < MT; ++mi) {
    #pragma unroll
    for (int t = 0; t < NT; ++t) {
      int col = colBase + wCol + 16 * t + nIdx;
      #pragma unroll
      for (int r = 0; r < 8; ++r) {
        int row = rowBase + wRow + 16 * mi + kh * 8 + r;
        float v = acc[mi][t][r];
        if (RELU) v = v > 0.f ? v : 0.f;
        if (OUT_BF16) ((__bf16*)C)[(size_t)row * ldc + col] = f2bf(v);
        else          ((float*) C)[(size_t)row * ldc + col] = v;
      }
    }
  }
}

// ---------------- asynchronous Hopfield retrieval ----------------
__global__ void hopfield_kernel(const float* __restrict__ enc,
                                const float* __restrict__ hop_w,
                                __bf16* __restrict__ comb)
{
  __shared__ float W[E_DIM * E_DIM];
  for (int i = threadIdx.x; i < E_DIM * E_DIM; i += 256) W[i] = hop_w[i];
  __syncthreads();

  int row = blockIdx.x * 256 + threadIdx.x;
  const float* e = enc + (size_t)row * E_DIM;
  uint32_t lo = 0, hi = 0;
  #pragma unroll
  for (int j = 0; j < 32; ++j) lo |= (e[j]      > 0.f ? 1u : 0u) << j;
  #pragma unroll
  for (int j = 0; j < 32; ++j) hi |= (e[32 + j] > 0.f ? 1u : 0u) << j;

  #pragma unroll 1
  for (int it = 0; it < NHOP_IT; ++it) {
    #pragma unroll 1
    for (int i = 0; i < E_DIM; ++i) {
      const float* wr = &W[i * E_DIM];
      float act = 0.f;
      #pragma unroll
      for (int j = 0; j < 32; ++j) { float w = wr[j];      act += ((lo >> j) & 1u) ? w : -w; }
      #pragma unroll
      for (int j = 0; j < 32; ++j) { float w = wr[32 + j]; act += ((hi >> j) & 1u) ? w : -w; }
      uint32_t s = act > 0.f ? 1u : 0u;
      if (i < 32) lo = (lo & ~(1u << i))        | (s << i);
      else        hi = (hi & ~(1u << (i - 32))) | (s << (i - 32));
    }
  }

  __bf16 one = f2bf(1.f), zero = f2bf(0.f);
  __bf16* dst = comb + (size_t)row * COMB_DIM + IN_DIM;
  #pragma unroll
  for (int j = 0; j < 32; ++j) dst[j]      = ((lo >> j) & 1u) ? one : zero;
  #pragma unroll
  for (int j = 0; j < 32; ++j) dst[32 + j] = ((hi >> j) & 1u) ? one : zero;
}

// ---------------- final 1024 -> 2 layer (wave-per-row GEMV) ----------------
__global__ void final_kernel(const __bf16* __restrict__ h,
                             const float* __restrict__ w3,   // [1024,2]
                             const float* __restrict__ b3,
                             float* __restrict__ out)
{
  int wave = threadIdx.x >> 5, lane = threadIdx.x & 31;
  int row = blockIdx.x * 8 + wave;
  const __bf16* hr = h + (size_t)row * HID_DIM;
  float a0 = 0.f, a1 = 0.f;
  for (int k = lane; k < HID_DIM; k += 32) {
    float hv = bf2f(hr[k]);
    a0 += hv * w3[k * 2 + 0];
    a1 += hv * w3[k * 2 + 1];
  }
  #pragma unroll
  for (int off = 16; off > 0; off >>= 1) {
    a0 += __shfl_down(a0, off);
    a1 += __shfl_down(a1, off);
  }
  if (lane == 0) {
    out[(size_t)row * 2 + 0] = a0 + b3[0];
    out[(size_t)row * 2 + 1] = a1 + b3[1];
  }
}

// ---------------- host launcher ----------------
extern "C" void kernel_launch(void* const* d_in, const int* in_sizes, int n_in,
                              void* d_out, int out_size, void* d_ws, size_t ws_size,
                              hipStream_t stream)
{
  const float* x     = (const float*)d_in[0];
  const float* e_w1  = (const float*)d_in[1];
  const float* e_b1  = (const float*)d_in[2];
  const float* e_w2  = (const float*)d_in[3];
  const float* e_b2  = (const float*)d_in[4];
  const float* hop_w = (const float*)d_in[5];
  const float* n_w1  = (const float*)d_in[6];
  const float* n_b1  = (const float*)d_in[7];
  const float* n_w2  = (const float*)d_in[8];
  const float* n_b2  = (const float*)d_in[9];
  const float* n_w3  = (const float*)d_in[10];
  const float* n_b3  = (const float*)d_in[11];
  float* out = (float*)d_out;

  char* ws = (char*)d_ws;
  size_t off = 0;
  auto wsAlloc = [&](size_t bytes) -> void* {
    void* p = ws + off;
    off = (off + bytes + 255) & ~(size_t)255;
    return p;
  };
  __bf16* combined = (__bf16*)wsAlloc((size_t)BATCH * COMB_DIM * 2);
  __bf16* bufA     = (__bf16*)wsAlloc((size_t)BATCH * HID_DIM * 2);
  __bf16* bufB     = (__bf16*)wsAlloc((size_t)BATCH * HID_DIM * 2);
  float*  enc      = (float*) wsAlloc((size_t)BATCH * E_DIM * 4);
  // weights pre-transposed to [N,K] bf16 so GEMM B-tiles DMA straight to LDS
  __bf16* w1t      = (__bf16*)wsAlloc((size_t)IN_DIM * HID_DIM * 2);    // [1024,512]
  __bf16* w2t      = (__bf16*)wsAlloc((size_t)HID_DIM * E_DIM * 2);     // [64,1024]
  __bf16* nw1t     = (__bf16*)wsAlloc((size_t)COMB_DIM * HID_DIM * 2);  // [1024,576]
  __bf16* nw2t     = (__bf16*)wsAlloc((size_t)HID_DIM * HID_DIM * 2);   // [1024,1024]

  int n;
  n = IN_DIM * HID_DIM;   cvt_t_kernel<<<(n + 255) / 256, 256, 0, stream>>>(e_w1, w1t, IN_DIM, HID_DIM);
  n = HID_DIM * E_DIM;    cvt_t_kernel<<<(n + 255) / 256, 256, 0, stream>>>(e_w2, w2t, HID_DIM, E_DIM);
  n = COMB_DIM * HID_DIM; cvt_t_kernel<<<(n + 255) / 256, 256, 0, stream>>>(n_w1, nw1t, COMB_DIM, HID_DIM);
  n = HID_DIM * HID_DIM;  cvt_t_kernel<<<(n + 255) / 256, 256, 0, stream>>>(n_w2, nw2t, HID_DIM, HID_DIM);
  n = BATCH * IN_DIM;     x_pack_kernel<<<(n + 255) / 256, 256, 0, stream>>>(x, combined);

  dim3 blk(256);
  // Encoder L1: relu(x @ e_w1 + b1) -> bufA [B,1024] bf16 (block 128x128)
  gemm_bf16_wmma<2, 4, 4, 2, true, true>
      <<<dim3(BATCH / 128, HID_DIM / 128), blk, 0, stream>>>(
      combined, COMB_DIM, w1t, IN_DIM, e_b1, bufA, HID_DIM, IN_DIM);
  // Encoder L2: bufA @ e_w2 + b2 -> enc [B,64] fp32 (block 128x64)
  gemm_bf16_wmma<1, 4, 8, 1, false, false>
      <<<dim3(BATCH / 128, E_DIM / 64), blk, 0, stream>>>(
      bufA, HID_DIM, w2t, HID_DIM, e_b2, enc, E_DIM, HID_DIM);
  // Hopfield retrieval -> combined cols 512..575
  hopfield_kernel<<<BATCH / 256, 256, 0, stream>>>(enc, hop_w, combined);
  // QNet L1: relu(combined @ n_w1 + b1) -> bufB
  gemm_bf16_wmma<2, 4, 4, 2, true, true>
      <<<dim3(BATCH / 128, HID_DIM / 128), blk, 0, stream>>>(
      combined, COMB_DIM, nw1t, COMB_DIM, n_b1, bufB, HID_DIM, COMB_DIM);
  // QNet L2: relu(bufB @ n_w2 + b2) -> bufA (reuse)
  gemm_bf16_wmma<2, 4, 4, 2, true, true>
      <<<dim3(BATCH / 128, HID_DIM / 128), blk, 0, stream>>>(
      bufB, HID_DIM, nw2t, HID_DIM, n_b2, bufA, HID_DIM, HID_DIM);
  // QNet L3: bufA @ n_w3 + b3 -> out [B,2]
  final_kernel<<<BATCH / 8, 256, 0, stream>>>(bufA, n_w3, n_b3, out);
}
